// MultiHeadAttention_3831110828829
// MI455X (gfx1250) — matrix operand order repaired and checked by reading, not compiled, not run on hardware
//
#include <hip/hip_runtime.h>
#include <math.h>

// Problem constants (must match reference)
#define B_     2
#define S_     2048
#define D_     512
#define H_     8
#define DEPTH_ 64
#define BS_    (B_ * S_)                  // 4096
#define BH_    (B_ * H_)                  // 16
#define NSD_   ((size_t)BH_ * S_ * DEPTH_)  // 2,097,152 elems per head-split tensor

typedef __attribute__((ext_vector_type(8)))  float   v8f;
typedef __attribute__((ext_vector_type(16))) __bf16  v16bf;
typedef __attribute__((ext_vector_type(8)))  __bf16  v8bf;
typedef __attribute__((ext_vector_type(4)))  __bf16  v4bf;

// D(16x16 f32) = A(16x32 bf16) * B(32x16 bf16) + C — CDNA5 V_WMMA_F32_16X16X32_BF16
static __device__ __forceinline__ v8f wmma_bf16(v16bf a, v16bf b, v8f c) {
    return __builtin_amdgcn_wmma_f32_16x16x32_bf16(
        /*neg_a=*/false, a, /*neg_b=*/false, b,
        /*c_mod=*/(short)0, c, /*reuse_a=*/false, /*reuse_b=*/false);
}

// Build a 16-elem bf16 fragment from two contiguous 8-elem (16 B) chunks.
// A-frag (16-bit, §7.12.2): elems 0..7 = K(8g..8g+7), elems 8..15 = K(16+8g..).
// B-frag: same K order as A-frag -> chunks p+8g, p+16+8g.
static __device__ __forceinline__ v16bf frag2(const __bf16* __restrict__ p0,
                                              const __bf16* __restrict__ p1) {
    v8bf lo = *(const v8bf*)p0;
    v8bf hi = *(const v8bf*)p1;
    v16bf r;
    #pragma unroll
    for (int i = 0; i < 8; ++i) { r[i] = lo[i]; r[i + 8] = hi[i]; }
    return r;
}

// A-frag assembled from f32 memory (attn rows), converting to bf16 in registers.
static __device__ __forceinline__ v16bf frag2_f32(const float* __restrict__ p0,
                                                  const float* __restrict__ p1) {
    float4 a0 = ((const float4*)p0)[0], a1 = ((const float4*)p0)[1];
    float4 b0 = ((const float4*)p1)[0], b1 = ((const float4*)p1)[1];
    v16bf r;
    r[0] = (__bf16)a0.x; r[1] = (__bf16)a0.y; r[2]  = (__bf16)a0.z; r[3]  = (__bf16)a0.w;
    r[4] = (__bf16)a1.x; r[5] = (__bf16)a1.y; r[6]  = (__bf16)a1.z; r[7]  = (__bf16)a1.w;
    r[8] = (__bf16)b0.x; r[9] = (__bf16)b0.y; r[10] = (__bf16)b0.z; r[11] = (__bf16)b0.w;
    r[12] = (__bf16)b1.x; r[13] = (__bf16)b1.y; r[14] = (__bf16)b1.z; r[15] = (__bf16)b1.w;
    return r;
}

static __device__ __forceinline__ float wave_max(float v) {
    #pragma unroll
    for (int off = 16; off; off >>= 1) v = fmaxf(v, __shfl_xor(v, off, 32));
    return v;
}
static __device__ __forceinline__ float wave_sum(float v) {
    #pragma unroll
    for (int off = 16; off; off >>= 1) v += __shfl_xor(v, off, 32);
    return v;
}

// ---------------------------------------------------------------------------
// Prep 1: f32 -> bf16 copy (4 elems/thread).
// ---------------------------------------------------------------------------
__global__ void __launch_bounds__(256)
cvt_bf16_kernel(const float* __restrict__ in, __bf16* __restrict__ out) {
    const int i = blockIdx.x * 256 + threadIdx.x;
    float4 v = ((const float4*)in)[i];
    v4bf o;
    o[0] = (__bf16)v.x; o[1] = (__bf16)v.y; o[2] = (__bf16)v.z; o[3] = (__bf16)v.w;
    ((v4bf*)out)[i] = o;
}

// ---------------------------------------------------------------------------
// Prep 2: weight transpose + convert: Wt[n][k] = (bf16)W[k][n], D x D.
// ---------------------------------------------------------------------------
__global__ void __launch_bounds__(256)
wtrans_kernel(const float* __restrict__ w, __bf16* __restrict__ wt) {
    const int idx = blockIdx.x * 256 + threadIdx.x;   // D*D threads
    const int k = idx >> 9;          // /D_
    const int n = idx & (D_ - 1);
    wt[(size_t)n * D_ + k] = (__bf16)w[idx];
}

// ---------------------------------------------------------------------------
// Kernel 1: fused Q/K/V projections (bf16 in, f32 accum, bf16 out).
// Q,K -> [B,H,S,64]; V -> transposed [B,H,64,S] (so attn@V B-frags are
// contiguous).  grid=(BS/16, D/128, 3), block=256.
// ---------------------------------------------------------------------------
__global__ void __launch_bounds__(256)
qkv_proj_kernel(const __bf16* __restrict__ xq, const __bf16* __restrict__ xk,
                const __bf16* __restrict__ xv,
                const __bf16* __restrict__ wtq, const __bf16* __restrict__ wtk,
                const __bf16* __restrict__ wtv,
                const float* __restrict__ bq, const float* __restrict__ bk,
                const float* __restrict__ bv,
                __bf16* __restrict__ qb, __bf16* __restrict__ kb,
                __bf16* __restrict__ vtb) {
    const int lane = threadIdx.x & 31;
    const int wave = threadIdx.x >> 5;
    const int z    = blockIdx.z;

    const __bf16* X    = (z == 0) ? xq  : (z == 1) ? xk  : xv;
    const __bf16* WT   = (z == 0) ? wtq : (z == 1) ? wtk : wtv;
    const float*  bias = (z == 0) ? bq  : (z == 1) ? bk  : bv;

    const int row0 = blockIdx.x * 16;
    const int col0 = blockIdx.y * 128 + wave * 16;

    const int ln = lane & 15;
    const int g8 = (lane >> 4) << 3;      // 0 or 8
    const int g16 = g8 << 1;              // 0 or 16
    const int mb = g8;                    // C/D M base (0 or 8)

    const __bf16* arow = X  + (size_t)(row0 + ln) * D_;
    const __bf16* brow = WT + (size_t)(col0 + ln) * D_;

    v8f c = {};
    #pragma unroll
    for (int k0 = 0; k0 < D_; k0 += 32) {
        v16bf a = frag2(arow + k0 + g8,  arow + k0 + 16 + g8);
        v16bf b = frag2(brow + k0 + g8,  brow + k0 + 16 + g8);
        c = wmma_bf16(a, b, c);
    }

    const int col  = col0 + ln;
    const int head = col >> 6;
    const int dd   = col & 63;
    const float bb = bias[col];
    #pragma unroll
    for (int r = 0; r < 8; ++r) {
        const int row = row0 + mb + r;
        const int bi  = row >> 11;        // /S_
        const int s   = row & (S_ - 1);
        const __bf16 val = (__bf16)(c[r] + bb);
        if (z == 2)
            vtb[(((size_t)bi * H_ + head) * DEPTH_ + dd) * S_ + s] = val;   // [B,H,64,S]
        else {
            __bf16* dst = (z == 0) ? qb : kb;
            dst[(((size_t)bi * H_ + head) * S_ + s) * DEPTH_ + dd] = val;   // [B,H,S,64]
        }
    }
}

// ---------------------------------------------------------------------------
// Kernel 2: logits = Q·K^T / 8 with causal mask folded in (-10000 masked).
// Two 16x16x32 WMMAs per tile (DEPTH=64).  K^T B-frags are contiguous in K's
// row-major depth.  grid=(S/16, S/128, B*H), block=256.
// ---------------------------------------------------------------------------
__global__ void __launch_bounds__(256)
logits_kernel(const __bf16* __restrict__ qb, const __bf16* __restrict__ kb,
              float* __restrict__ attn) {
    const int lane = threadIdx.x & 31;
    const int wave = threadIdx.x >> 5;
    const int bh   = blockIdx.z;

    const __bf16* Q = qb + (size_t)bh * S_ * DEPTH_;
    const __bf16* K = kb + (size_t)bh * S_ * DEPTH_;
    float* A = attn + (size_t)bh * S_ * S_;

    const int row0 = blockIdx.x * 16;
    const int col0 = blockIdx.y * 128 + wave * 16;

    const int ln  = lane & 15;
    const int g8  = (lane >> 4) << 3;
    const int g16 = g8 << 1;
    const int mb  = g8;

    if (col0 > row0 + 15) {               // tile entirely above the diagonal
        #pragma unroll
        for (int r = 0; r < 8; ++r)
            A[(size_t)(row0 + mb + r) * S_ + col0 + ln] = -10000.0f;
        return;
    }

    const __bf16* arow = Q + (size_t)(row0 + ln) * DEPTH_;
    const __bf16* brow = K + (size_t)(col0 + ln) * DEPTH_;

    v8f c = {};
    #pragma unroll
    for (int d0 = 0; d0 < DEPTH_; d0 += 32) {
        v16bf a = frag2(arow + d0 + g8,  arow + d0 + 16 + g8);
        v16bf b = frag2(brow + d0 + g8,  brow + d0 + 16 + g8);
        c = wmma_bf16(a, b, c);
    }
    #pragma unroll
    for (int r = 0; r < 8; ++r) {
        const int qi = row0 + mb + r;
        const int kj = col0 + ln;
        const float v = (kj <= qi) ? c[r] * 0.125f : -10000.0f;   // 1/sqrt(64)
        A[(size_t)qi * S_ + kj] = v;
    }
}

// ---------------------------------------------------------------------------
// Kernel 3: in-place f32 row softmax over S=2048.  grid=(S, B*H), block=256.
// ---------------------------------------------------------------------------
__global__ void __launch_bounds__(256)
softmax_kernel(float* __restrict__ attn) {
    const int tid  = threadIdx.x;
    const int lane = tid & 31;
    const int wv   = tid >> 5;
    float* row = attn + ((size_t)blockIdx.y * S_ + blockIdx.x) * S_;

    float4 a = ((const float4*)row)[tid];
    float4 b = ((const float4*)row)[256 + tid];

    __shared__ float smax[8];
    __shared__ float ssum[8];

    float m = fmaxf(fmaxf(fmaxf(a.x, a.y), fmaxf(a.z, a.w)),
                    fmaxf(fmaxf(b.x, b.y), fmaxf(b.z, b.w)));
    m = wave_max(m);
    if (lane == 0) smax[wv] = m;
    __syncthreads();
    if (tid == 0) {
        float mm = smax[0];
        #pragma unroll
        for (int i = 1; i < 8; ++i) mm = fmaxf(mm, smax[i]);
        smax[0] = mm;
    }
    __syncthreads();
    m = smax[0];

    a.x = __expf(a.x - m); a.y = __expf(a.y - m);
    a.z = __expf(a.z - m); a.w = __expf(a.w - m);
    b.x = __expf(b.x - m); b.y = __expf(b.y - m);
    b.z = __expf(b.z - m); b.w = __expf(b.w - m);

    float s = (a.x + a.y + a.z + a.w) + (b.x + b.y + b.z + b.w);
    s = wave_sum(s);
    if (lane == 0) ssum[wv] = s;
    __syncthreads();
    if (tid == 0) {
        float t = 0.f;
        #pragma unroll
        for (int i = 0; i < 8; ++i) t += ssum[i];
        ssum[0] = t;
    }
    __syncthreads();
    const float inv = 1.0f / ssum[0];     // >= exp(0)=1, never zero

    a.x *= inv; a.y *= inv; a.z *= inv; a.w *= inv;
    b.x *= inv; b.y *= inv; b.z *= inv; b.w *= inv;
    ((float4*)row)[tid]       = a;
    ((float4*)row)[256 + tid] = b;
}

// ---------------------------------------------------------------------------
// Kernel 4: ctx = attn @ V.  attn read as f32, converted to bf16 in registers;
// V is head-transposed so B-frags are contiguous.  Causal early termination.
// grid=(S/32, 1, B*H), block=256 (8 waves: 2 q-tiles x 4 n-tiles).
// Output: bf16 merged-head [B,S,D] (feeds the output-projection GEMM).
// ---------------------------------------------------------------------------
__global__ void __launch_bounds__(256)
ctx_kernel(const float* __restrict__ attn, const __bf16* __restrict__ vtb,
           __bf16* __restrict__ ctxb) {
    const int lane = threadIdx.x & 31;
    const int wave = threadIdx.x >> 5;
    const int bh   = blockIdx.z;
    const int b    = bh >> 3;
    const int h    = bh & 7;

    const float*  A  = attn + (size_t)bh * S_ * S_;
    const __bf16* Vt = vtb + (size_t)bh * DEPTH_ * S_;

    const int row0 = blockIdx.x * 32 + (wave >> 2) * 16;
    const int col0 = (wave & 3) * 16;

    const int ln  = lane & 15;
    const int g8  = (lane >> 4) << 3;
    const int g16 = g8 << 1;
    const int mb  = g8;

    const float*  arow = A  + (size_t)(row0 + ln) * S_;
    const __bf16* brow = Vt + (size_t)(col0 + ln) * S_;

    v8f c = {};
    const int kmax = ((row0 >> 5) + 1) << 5;   // attn == 0 beyond diagonal
    #pragma unroll 2
    for (int k0 = 0; k0 < kmax; k0 += 32) {
        if ((k0 & 127) == 0)                    // stream attn rows ahead
            __builtin_prefetch(arow + k0 + 128, 0, 0);
        v16bf a = frag2_f32(arow + k0 + g8,  arow + k0 + 16 + g8);
        v16bf b = frag2(brow + k0 + g8,  brow + k0 + 16 + g8);
        c = wmma_bf16(a, b, c);
    }
    #pragma unroll
    for (int r = 0; r < 8; ++r) {
        const int qi = row0 + mb + r;
        ctxb[((size_t)b * S_ + qi) * D_ + h * DEPTH_ + col0 + ln] = (__bf16)c[r];
    }
}

// ---------------------------------------------------------------------------
// Kernel 5: proj = ctx @ wo + bo + residual(query), f32 out.
// grid=(BS/16, D/128), block=256.
// ---------------------------------------------------------------------------
__global__ void __launch_bounds__(256)
outproj_kernel(const __bf16* __restrict__ ctxb, const __bf16* __restrict__ wto,
               const float* __restrict__ bo, const float* __restrict__ q_in,
               float* __restrict__ proj) {
    const int lane = threadIdx.x & 31;
    const int wave = threadIdx.x >> 5;

    const int row0 = blockIdx.x * 16;
    const int col0 = blockIdx.y * 128 + wave * 16;

    const int ln  = lane & 15;
    const int g8  = (lane >> 4) << 3;
    const int g16 = g8 << 1;
    const int mb  = g8;

    const __bf16* arow = ctxb + (size_t)(row0 + ln) * D_;
    const __bf16* brow = wto + (size_t)(col0 + ln) * D_;

    v8f c = {};
    #pragma unroll
    for (int k0 = 0; k0 < D_; k0 += 32) {
        v16bf a = frag2(arow + k0 + g8,  arow + k0 + 16 + g8);
        v16bf b = frag2(brow + k0 + g8,  brow + k0 + 16 + g8);
        c = wmma_bf16(a, b, c);
    }
    const int col  = col0 + ln;
    const float bb = bo[col];
    #pragma unroll
    for (int r = 0; r < 8; ++r) {
        const size_t idx = (size_t)(row0 + mb + r) * D_ + col;
        proj[idx] = c[r] + bb + q_in[idx];
    }
}

// ---------------------------------------------------------------------------
// Kernel 6: LayerNorm over D=512 per row.  grid=BS, block=256.
// ---------------------------------------------------------------------------
__global__ void __launch_bounds__(256)
layernorm_kernel(const float* __restrict__ proj, const float* __restrict__ gamma,
                 const float* __restrict__ beta, float* __restrict__ out) {
    const int tid  = threadIdx.x;
    const int lane = tid & 31;
    const int wv   = tid >> 5;
    const float* x = proj + (size_t)blockIdx.x * D_;

    float2 v = ((const float2*)x)[tid];
    float s  = v.x + v.y;
    float s2 = v.x * v.x + v.y * v.y;
    s  = wave_sum(s);
    s2 = wave_sum(s2);

    __shared__ float r0[8];
    __shared__ float r1[8];
    if (lane == 0) { r0[wv] = s; r1[wv] = s2; }
    __syncthreads();
    if (tid == 0) {
        float t0 = 0.f, t1 = 0.f;
        #pragma unroll
        for (int i = 0; i < 8; ++i) { t0 += r0[i]; t1 += r1[i]; }
        r0[0] = t0; r1[0] = t1;
    }
    __syncthreads();
    const float mean = r0[0] * (1.0f / D_);
    const float var  = r1[0] * (1.0f / D_) - mean * mean;
    const float rs   = rsqrtf(var + 1e-6f);

    const int c0 = tid * 2;
    float* o = out + (size_t)blockIdx.x * D_;
    o[c0]     = (v.x - mean) * rs * gamma[c0]     + beta[c0];
    o[c0 + 1] = (v.y - mean) * rs * gamma[c0 + 1] + beta[c0 + 1];
}

// ---------------------------------------------------------------------------
extern "C" void kernel_launch(void* const* d_in, const int* in_sizes, int n_in,
                              void* d_out, int out_size, void* d_ws, size_t ws_size,
                              hipStream_t stream) {
    const float* query = (const float*)d_in[0];
    const float* key   = (const float*)d_in[1];
    const float* value = (const float*)d_in[2];
    // d_in[3] is the causal mask; folded analytically into logits_kernel.
    const float* wq = (const float*)d_in[4];
    const float* bq = (const float*)d_in[5];
    const float* wk = (const float*)d_in[6];
    const float* bk = (const float*)d_in[7];
    const float* wv = (const float*)d_in[8];
    const float* bv = (const float*)d_in[9];
    const float* wo = (const float*)d_in[10];
    const float* bo = (const float*)d_in[11];
    const float* gamma = (const float*)d_in[12];
    const float* beta  = (const float*)d_in[13];

    float* out0 = (float*)d_out;                 // [B,S,D]
    float* attn = out0 + (size_t)BS_ * D_;       // [B,H,S,S]

    // Workspace carve-up (all offsets are multiples of 16 B).
    __bf16* xq  = (__bf16*)d_ws;                 // [BS,D] bf16 input copies
    __bf16* xk  = xq  + (size_t)BS_ * D_;
    __bf16* xv  = xk  + (size_t)BS_ * D_;
    __bf16* wtq = xv  + (size_t)BS_ * D_;        // [D,D] bf16 transposed weights
    __bf16* wtk = wtq + (size_t)D_ * D_;
    __bf16* wtv = wtk + (size_t)D_ * D_;
    __bf16* wto = wtv + (size_t)D_ * D_;
    __bf16* qb  = wto + (size_t)D_ * D_;         // [B,H,S,64]
    __bf16* kb  = qb  + NSD_;                    // [B,H,S,64]
    __bf16* vtb = kb  + NSD_;                    // [B,H,64,S]
    __bf16* ctxb = vtb + NSD_;                   // [B,S,D]
    float*  proj = (float*)(ctxb + NSD_);        // [BS,D] f32  (total ~38 MB)

    // Prep: bf16 input copies + transposed bf16 weights.
    const int cvtBlocks = (BS_ * D_ / 4) / 256;  // 2048
    cvt_bf16_kernel<<<cvtBlocks, 256, 0, stream>>>(query, xq);
    cvt_bf16_kernel<<<cvtBlocks, 256, 0, stream>>>(key,   xk);
    cvt_bf16_kernel<<<cvtBlocks, 256, 0, stream>>>(value, xv);
    const int wtBlocks = (D_ * D_) / 256;        // 1024
    wtrans_kernel<<<wtBlocks, 256, 0, stream>>>(wq, wtq);
    wtrans_kernel<<<wtBlocks, 256, 0, stream>>>(wk, wtk);
    wtrans_kernel<<<wtBlocks, 256, 0, stream>>>(wv, wtv);
    wtrans_kernel<<<wtBlocks, 256, 0, stream>>>(wo, wto);

    qkv_proj_kernel<<<dim3(BS_ / 16, D_ / 128, 3), 256, 0, stream>>>(
        xq, xk, xv, wtq, wtk, wtv, bq, bk, bv, qb, kb, vtb);
    logits_kernel<<<dim3(S_ / 16, S_ / 128, BH_), 256, 0, stream>>>(qb, kb, attn);
    softmax_kernel<<<dim3(S_, BH_), 256, 0, stream>>>(attn);
    ctx_kernel<<<dim3(S_ / 32, 1, BH_), 256, 0, stream>>>(attn, vtb, ctxb);
    outproj_kernel<<<dim3(BS_ / 16, D_ / 128), 256, 0, stream>>>(
        ctxb, wto, bo, query, proj);
    layernorm_kernel<<<BS_, 256, 0, stream>>>(proj, gamma, beta, out0);
}
